// LSTMBase_39384850104646
// MI455X (gfx1250) — compile-verified, matrix operand order checked by reading, not run on hardware
//
#include <hip/hip_runtime.h>
#include <hip/hip_bf16.h>

// ---------------------------------------------------------------------------
// Problem dims
// ---------------------------------------------------------------------------
#define DD   128
#define TT   256
#define NN   512
#define AA   16
#define HH   256
#define EMBD 512
#define TN   (TT * NN)

typedef __bf16 bf16_t;
typedef __attribute__((ext_vector_type(16))) __bf16 bf16x16;
typedef __attribute__((ext_vector_type(8)))  float  floatx8;

// ---------------------------------------------------------------------------
// WMMA helpers (CDNA5 wave32, 16x16x32 bf16 -> f32)
// ---------------------------------------------------------------------------
__device__ __forceinline__ floatx8 wmma_bf16(bf16x16 a, bf16x16 b, floatx8 c) {
    // (neg_a, A, neg_b, B, c_mod, C, reuse_a, reuse_b)
    return __builtin_amdgcn_wmma_f32_16x16x32_bf16(false, a, false, b,
                                                   (short)0, c, false, false);
}

// Load one 16-lane-half fragment: two contiguous 16B chunks (K base..+7, K base+16..+23)
__device__ __forceinline__ bf16x16 load_frag(const bf16_t* p) {
    union { uint4 u[2]; bf16x16 v; } f;
    f.u[0] = *reinterpret_cast<const uint4*>(p);
    f.u[1] = *reinterpret_cast<const uint4*>(p + 16);
    return f.v;
}

// ---------------------------------------------------------------------------
// Fast transcendentals: CDNA5 has a native v_tanh_f32 TRANS op.
// sigmoid(x) = 0.5*tanh(x/2) + 0.5  -> one TRANS + one FMA (no IEEE divide).
// ---------------------------------------------------------------------------
#if __has_builtin(__builtin_amdgcn_tanhf)
__device__ __forceinline__ float fast_tanh(float x) { return __builtin_amdgcn_tanhf(x); }
#elif __has_builtin(__builtin_amdgcn_tanh_f32)
__device__ __forceinline__ float fast_tanh(float x) { return __builtin_amdgcn_tanh_f32(x); }
#else
__device__ __forceinline__ float fast_tanh(float x) { return tanhf(x); }
#endif
__device__ __forceinline__ float sigm(float x) { return fmaf(0.5f, fast_tanh(0.5f * x), 0.5f); }
__device__ __forceinline__ float silu(float x) { return x * sigm(x); }

// One 16x16 output tile, K-loop over kdim. A in LDS/global bf16 (row-major,
// stride astride halfs), B = transposed weight Wt[out][in] (stride kdim).
__device__ __forceinline__ floatx8 gemm_tile(const bf16_t* abuf, int astride,
                                             const bf16_t* wt, int kdim,
                                             int coltile, int lane) {
    floatx8 acc = {0.f,0.f,0.f,0.f,0.f,0.f,0.f,0.f};
    const int lrow = lane & 15;
    const int kh   = lane >> 4;
    const bf16_t* arow = abuf + lrow * astride + kh * 8;
    const bf16_t* wrow = wt + (size_t)(coltile * 16 + lrow) * kdim + kh * 8;
#pragma unroll 4
    for (int k0 = 0; k0 < kdim; k0 += 32) {
        acc = wmma_bf16(load_frag(arow + k0), load_frag(wrow + k0), acc);
    }
    return acc;
}

// ---------------------------------------------------------------------------
// Weight prep: fp32 W[in][out] -> bf16 Wt[out][in]. All dims are powers of 2.
// ---------------------------------------------------------------------------
__global__ void transpose_bf16_kernel(const float* __restrict__ W,
                                      bf16_t* __restrict__ Wt,
                                      int log2in, int out) {
    const int in = 1 << log2in;
    int i = blockIdx.x * blockDim.x + threadIdx.x;
    if (i < in * out) {
        int o = i >> log2in, k = i & (in - 1);
        Wt[(size_t)o * in + k] = (bf16_t)W[(size_t)k * out + o];
    }
}

__global__ void init_state_kernel(const float* __restrict__ h0,
                                  const float* __restrict__ c0,
                                  float* __restrict__ hdst,
                                  float* __restrict__ cdst, int n) {
    int i = blockIdx.x * blockDim.x + threadIdx.x;
    if (i < n) { hdst[i] = h0[i]; cdst[i] = c0[i]; }
}

// ---------------------------------------------------------------------------
// LayerNorm over 16 rows x L cols held in LDS fp32; writes fp32 + bf16 copies
// ---------------------------------------------------------------------------
__device__ void ln16(float (*fbuf)[520], bf16_t (*abuf)[528],
                     float (*redA)[16], float (*redB)[16],
                     const float* __restrict__ g, const float* __restrict__ b,
                     int L, float invL, int tid) {
    const int row = tid >> 4, sub = tid & 15;
    float s = 0.f, s2 = 0.f;
    for (int c = sub; c < L; c += 16) { float v = fbuf[row][c]; s += v; s2 += v * v; }
    redA[row][sub] = s; redB[row][sub] = s2;
    __syncthreads();
    if (sub == 0) {
        float m = 0.f, q = 0.f;
#pragma unroll
        for (int i = 0; i < 16; i++) { m += redA[row][i]; q += redB[row][i]; }
        m *= invL; q = q * invL - m * m;
        redA[row][0] = m;
        redB[row][0] = rsqrtf(q + 1e-5f);
    }
    __syncthreads();
    const float m = redA[row][0], r = redB[row][0];
    for (int c = sub; c < L; c += 16) {
        float v = (fbuf[row][c] - m) * r * g[c] + b[c];
        fbuf[row][c] = v;
        abuf[row][c] = (bf16_t)v;
    }
    __syncthreads();
}

// ---------------------------------------------------------------------------
// Fused embed MLP: 16 rows/block, 8 waves, full chain LN0..fc -> core_in bf16
// ---------------------------------------------------------------------------
__global__ __launch_bounds__(256) void embed_kernel(
    const float* __restrict__ obs,
    const float* __restrict__ ln0_g, const float* __restrict__ ln0_b,
    const float* __restrict__ ln1_g, const float* __restrict__ ln1_b,
    const float* __restrict__ ln2_g, const float* __restrict__ ln2_b,
    const float* __restrict__ ln3_g, const float* __restrict__ ln3_b,
    const float* __restrict__ ln4_g, const float* __restrict__ ln4_b,
    const float* __restrict__ fc1_b, const float* __restrict__ fc2_b,
    const float* __restrict__ fc3_b, const float* __restrict__ fc4_b,
    const float* __restrict__ fc_b,
    const bf16_t* __restrict__ wt_fc1, const bf16_t* __restrict__ wt_fc2,
    const bf16_t* __restrict__ wt_fc3, const bf16_t* __restrict__ wt_fc4,
    const bf16_t* __restrict__ wt_fc,
    bf16_t* __restrict__ core_in) {
    __shared__ __align__(16) float  fbuf[16][520];
    __shared__ __align__(16) bf16_t abuf[16][528];
    __shared__ float redA[16][16];
    __shared__ float redB[16][16];

    const int tid  = threadIdx.x;
    const int wave = tid >> 5;
    const int lane = tid & 31;
    const int lrow = lane & 15, kh = lane >> 4, r0 = kh * 8;
    const size_t rowg0 = (size_t)blockIdx.x * 16;

    // obs -> LDS fp32
    for (int i = tid; i < 16 * DD; i += 256) {
        int r = i >> 7, c = i & 127;
        fbuf[r][c] = obs[(rowg0 + r) * DD + c];
    }
    __syncthreads();
    ln16(fbuf, abuf, redA, redB, ln0_g, ln0_b, DD, 1.f / DD, tid);

    // fc1 (128->128) + SiLU + residual, then LN1
    {
        floatx8 acc = gemm_tile(&abuf[0][0], 528, wt_fc1, 128, wave, lane);
        const int col = wave * 16 + lrow;
        const float bias = fc1_b[col];
#pragma unroll
        for (int r = 0; r < 8; r++) fbuf[r0 + r][col] += silu(acc[r] + bias);
    }
    __syncthreads();
    ln16(fbuf, abuf, redA, redB, ln1_g, ln1_b, DD, 1.f / DD, tid);

    // concat: copy x1 to cols [128,256)
    for (int i = tid; i < 16 * 128; i += 256) {
        int r = i >> 7, c = i & 127; fbuf[r][128 + c] = fbuf[r][c];
    }
    __syncthreads();
    // fc2 (128->128) SiLU into cols [0,128), then LN2 over 256
    {
        floatx8 acc = gemm_tile(&abuf[0][0], 528, wt_fc2, 128, wave, lane);
        const int col = wave * 16 + lrow;
        const float bias = fc2_b[col];
#pragma unroll
        for (int r = 0; r < 8; r++) fbuf[r0 + r][col] = silu(acc[r] + bias);
    }
    __syncthreads();
    ln16(fbuf, abuf, redA, redB, ln2_g, ln2_b, 256, 1.f / 256.f, tid);

    // concat: copy x2 to cols [256,512)
    for (int i = tid; i < 16 * 256; i += 256) {
        int r = i >> 8, c = i & 255; fbuf[r][256 + c] = fbuf[r][c];
    }
    __syncthreads();
    // fc3 (256->256) SiLU into cols [0,256), then LN3 over 512
    for (int tt = wave; tt < 16; tt += 8) {
        floatx8 acc = gemm_tile(&abuf[0][0], 528, wt_fc3, 256, tt, lane);
        const int col = tt * 16 + lrow;
        const float bias = fc3_b[col];
#pragma unroll
        for (int r = 0; r < 8; r++) fbuf[r0 + r][col] = silu(acc[r] + bias);
    }
    __syncthreads();
    ln16(fbuf, abuf, redA, redB, ln3_g, ln3_b, EMBD, 1.f / EMBD, tid);

    // fc4 (512->512) SiLU, then LN4
    {
        floatx8 res[4];
#pragma unroll
        for (int j = 0; j < 4; j++)
            res[j] = gemm_tile(&abuf[0][0], 528, wt_fc4, 512, wave + 8 * j, lane);
        __syncthreads();   // done reading abuf before fbuf rewrite feeds LN4
#pragma unroll
        for (int j = 0; j < 4; j++) {
            const int col = (wave + 8 * j) * 16 + lrow;
            const float bias = fc4_b[col];
#pragma unroll
            for (int r = 0; r < 8; r++) fbuf[r0 + r][col] = silu(res[j][r] + bias);
        }
    }
    __syncthreads();
    ln16(fbuf, abuf, redA, redB, ln4_g, ln4_b, EMBD, 1.f / EMBD, tid);

    // fc (512->256) + ReLU -> core_in (bf16, global)
    for (int tt = wave; tt < 16; tt += 8) {
        floatx8 acc = gemm_tile(&abuf[0][0], 528, wt_fc, 512, tt, lane);
        const int col = tt * 16 + lrow;
        const float bias = fc_b[col];
#pragma unroll
        for (int r = 0; r < 8; r++) {
            float v = fmaxf(acc[r] + bias, 0.f);
            core_in[(rowg0 + r0 + r) * HH + col] = (bf16_t)v;
        }
    }
}

// ---------------------------------------------------------------------------
// One LSTM layer for one time step. Block = 16 rows x all 1024 gate columns.
// gates = xin@Wih^T + (h*mask)@Whh^T + b ; cell update fully in-register.
// ---------------------------------------------------------------------------
__global__ __launch_bounds__(256) void lstm_layer_kernel(
    const bf16_t* __restrict__ xin,      // [N rows at this step][256] bf16
    const float*  __restrict__ masks,    // [N] for this step
    float* __restrict__ hstate,          // [N][256] fp32 (in d_out h_n region)
    float* __restrict__ cstate,          // [N][256] fp32 (in d_out c_n region)
    const bf16_t* __restrict__ wt_ih,    // [1024][256] bf16
    const bf16_t* __restrict__ wt_hh,    // [1024][256] bf16
    const float* __restrict__ b_ih, const float* __restrict__ b_hh,
    bf16_t* __restrict__ hbf_out) {      // bf16 copy of new h
    __shared__ __align__(16) bf16_t hbuf[16][264];
    __shared__ float msk[16];

    const int tid = threadIdx.x;
    const int rowbase = blockIdx.x * 16;
    if (tid < 16) msk[tid] = masks[rowbase + tid];
    __syncthreads();
    // stage masked previous h as bf16 in LDS
    for (int i = tid; i < 16 * HH; i += 256) {
        int r = i >> 8, c = i & 255;
        hbuf[r][c] = (bf16_t)(hstate[(size_t)(rowbase + r) * HH + c] * msk[r]);
    }
    __syncthreads();

    const int wave = tid >> 5, lane = tid & 31;
    const int lrow = lane & 15, kh = lane >> 4;

    floatx8 acc[8];
#pragma unroll
    for (int j = 0; j < 8; j++) acc[j] = (floatx8){0.f,0.f,0.f,0.f,0.f,0.f,0.f,0.f};

    const bf16_t* xrow = xin + (size_t)(rowbase + lrow) * HH + kh * 8;
    const bf16_t* hrow = &hbuf[lrow][kh * 8];
#pragma unroll 2
    for (int k0 = 0; k0 < HH; k0 += 32) {
        bf16x16 xa = load_frag(xrow + k0);
        bf16x16 ha = load_frag(hrow + k0);
#pragma unroll
        for (int j = 0; j < 8; j++) {
            const size_t wr = (size_t)((wave + 8 * j) * 16 + lrow) * HH + k0 + kh * 8;
            acc[j] = wmma_bf16(xa, load_frag(wt_ih + wr), acc[j]);
            acc[j] = wmma_bf16(ha, load_frag(wt_hh + wr), acc[j]);
        }
    }
    // biases
#pragma unroll
    for (int j = 0; j < 8; j++) {
        const int g = wave * 16 + 128 * j + lrow;
        const float bb = b_ih[g] + b_hh[g];
#pragma unroll
        for (int r = 0; r < 8; r++) acc[j][r] += bb;
    }
    // cell update: i/f/g/o quarters live at acc[j], acc[j+2], acc[j+4], acc[j+6]
#pragma unroll
    for (int j = 0; j < 2; j++) {
#pragma unroll
        for (int r = 0; r < 8; r++) {
            const int row = rowbase + kh * 8 + r;
            const int col = wave * 16 + 128 * j + lrow;
            const float ig = acc[j][r], fg = acc[j + 2][r];
            const float gg = acc[j + 4][r], og = acc[j + 6][r];
            const float cold = cstate[(size_t)row * HH + col] * msk[kh * 8 + r];
            const float cn = sigm(fg) * cold + sigm(ig) * fast_tanh(gg);
            const float hn = sigm(og) * fast_tanh(cn);
            cstate[(size_t)row * HH + col] = cn;
            hstate[(size_t)row * HH + col] = hn;
            hbf_out[(size_t)row * HH + col] = (bf16_t)hn;
        }
    }
}

// ---------------------------------------------------------------------------
// Heads: fused actor1||critic1 (256->128) WMMA GEMM + ReLU, then tiny level-2
// ---------------------------------------------------------------------------
__global__ __launch_bounds__(256) void head_kernel(
    const bf16_t* __restrict__ core_out,       // [TN][256] bf16
    const bf16_t* __restrict__ wt_a1,          // [64][256]
    const bf16_t* __restrict__ wt_c1,          // [64][256]
    const float* __restrict__ a1_b, const float* __restrict__ c1_b,
    const float* __restrict__ a2_w, const float* __restrict__ a2_b,
    const float* __restrict__ c2_w, const float* __restrict__ c2_b,
    float* __restrict__ logits, float* __restrict__ value) {
    __shared__ float hid[16][136];
    const int tid = threadIdx.x;
    const int wave = tid >> 5, lane = tid & 31;
    const int lrow = lane & 15, kh = lane >> 4;
    const size_t rowg0 = (size_t)blockIdx.x * 16;

    const bf16_t* wt = (wave < 4) ? wt_a1 + (size_t)(wave * 16 + lrow) * HH
                                  : wt_c1 + (size_t)((wave - 4) * 16 + lrow) * HH;
    const bf16_t* xrow = core_out + (rowg0 + lrow) * HH + kh * 8;

    floatx8 acc = {0.f,0.f,0.f,0.f,0.f,0.f,0.f,0.f};
#pragma unroll
    for (int k0 = 0; k0 < HH; k0 += 32)
        acc = wmma_bf16(load_frag(xrow + k0), load_frag(wt + k0 + kh * 8), acc);

    const int col = wave * 16 + lrow;
    const float bias = (col < 64) ? a1_b[col] : c1_b[col - 64];
#pragma unroll
    for (int r = 0; r < 8; r++) hid[kh * 8 + r][col] = fmaxf(acc[r] + bias, 0.f);
    __syncthreads();

    // actor2: 64 -> 16 per row (16 rows x 16 outputs == 256 threads)
    const int row = tid >> 4, a = tid & 15;
    float s = a2_b[a];
#pragma unroll 8
    for (int k = 0; k < 64; k++) s += hid[row][k] * a2_w[k * AA + a];
    logits[(rowg0 + row) * AA + a] = s;

    // critic2: 64 -> 1 per row
    if (tid < 16) {
        float s2 = c2_b[0];
#pragma unroll 8
        for (int k = 0; k < 64; k++) s2 += hid[tid][64 + k] * c2_w[k];
        value[rowg0 + tid] = s2;
    }
}

// ---------------------------------------------------------------------------
// Host-side orchestration
// ---------------------------------------------------------------------------
extern "C" void kernel_launch(void* const* d_in, const int* in_sizes, int n_in,
                              void* d_out, int out_size, void* d_ws, size_t ws_size,
                              hipStream_t stream) {
    const float* obs    = (const float*)d_in[0];
    const float* masks  = (const float*)d_in[1];
    const float* h0     = (const float*)d_in[2];
    const float* c0     = (const float*)d_in[3];
    const float* ln0_g  = (const float*)d_in[4];
    const float* ln0_b  = (const float*)d_in[5];
    const float* fc1_w  = (const float*)d_in[6];
    const float* fc1_b  = (const float*)d_in[7];
    const float* ln1_g  = (const float*)d_in[8];
    const float* ln1_b  = (const float*)d_in[9];
    const float* fc2_w  = (const float*)d_in[10];
    const float* fc2_b  = (const float*)d_in[11];
    const float* ln2_g  = (const float*)d_in[12];
    const float* ln2_b  = (const float*)d_in[13];
    const float* fc3_w  = (const float*)d_in[14];
    const float* fc3_b  = (const float*)d_in[15];
    const float* ln3_g  = (const float*)d_in[16];
    const float* ln3_b  = (const float*)d_in[17];
    const float* fc4_w  = (const float*)d_in[18];
    const float* fc4_b  = (const float*)d_in[19];
    const float* ln4_g  = (const float*)d_in[20];
    const float* ln4_b  = (const float*)d_in[21];
    const float* fc_w   = (const float*)d_in[22];
    const float* fc_b   = (const float*)d_in[23];
    const float* w_ih   = (const float*)d_in[24];   // [2][256][1024]
    const float* w_hh   = (const float*)d_in[25];
    const float* b_ih   = (const float*)d_in[26];   // [2][1024]
    const float* b_hh   = (const float*)d_in[27];
    const float* a1_w   = (const float*)d_in[28];
    const float* a1_b   = (const float*)d_in[29];
    const float* a2_w   = (const float*)d_in[30];
    const float* a2_b   = (const float*)d_in[31];
    const float* c1_w   = (const float*)d_in[32];
    const float* c1_b   = (const float*)d_in[33];
    const float* c2_w   = (const float*)d_in[34];
    const float* c2_b   = (const float*)d_in[35];
    (void)in_sizes; (void)n_in; (void)out_size; (void)ws_size;

    // ---- d_out layout: logits | value | h_n[2] | c_n[2]
    float* out    = (float*)d_out;
    float* logits = out;
    float* value  = out + (size_t)TN * AA;
    float* h1     = value + TN;                 // h_n layer 0
    float* h2     = h1 + (size_t)NN * HH;       // h_n layer 1
    float* c1     = h2 + (size_t)NN * HH;       // c_n layer 0
    float* c2     = c1 + (size_t)NN * HH;       // c_n layer 1

    // ---- d_ws layout (bf16 unless noted)
    char* ws = (char*)d_ws;
    size_t off = 0;
    auto alloc = [&](size_t bytes) { size_t r = off; off += (bytes + 255) & ~(size_t)255; return r; };
    bf16_t* wt_fc1 = (bf16_t*)(ws + alloc((size_t)128 * 128 * 2));
    bf16_t* wt_fc2 = (bf16_t*)(ws + alloc((size_t)128 * 128 * 2));
    bf16_t* wt_fc3 = (bf16_t*)(ws + alloc((size_t)256 * 256 * 2));
    bf16_t* wt_fc4 = (bf16_t*)(ws + alloc((size_t)512 * 512 * 2));
    bf16_t* wt_fc  = (bf16_t*)(ws + alloc((size_t)256 * 512 * 2));
    bf16_t* wt_ih1 = (bf16_t*)(ws + alloc((size_t)1024 * 256 * 2));
    bf16_t* wt_ih2 = (bf16_t*)(ws + alloc((size_t)1024 * 256 * 2));
    bf16_t* wt_hh1 = (bf16_t*)(ws + alloc((size_t)1024 * 256 * 2));
    bf16_t* wt_hh2 = (bf16_t*)(ws + alloc((size_t)1024 * 256 * 2));
    bf16_t* wt_a1  = (bf16_t*)(ws + alloc((size_t)64 * 256 * 2));
    bf16_t* wt_c1  = (bf16_t*)(ws + alloc((size_t)64 * 256 * 2));
    bf16_t* core_in  = (bf16_t*)(ws + alloc((size_t)TN * HH * 2));
    bf16_t* h1bf     = (bf16_t*)(ws + alloc((size_t)NN * HH * 2));
    bf16_t* core_out = (bf16_t*)(ws + alloc((size_t)TN * HH * 2));

    // ---- state init (h0/c0 -> h_n/c_n regions of d_out)
    {
        const int n = 2 * NN * HH;
        init_state_kernel<<<(n + 255) / 256, 256, 0, stream>>>(h0, c0, h1, c1, n);
    }
    // ---- weight prep (fp32 -> transposed bf16); log2in since all in-dims are pow2
    auto tp = [&](const float* W, bf16_t* Wt, int log2in, int outd) {
        transpose_bf16_kernel<<<(((1 << log2in) * outd) + 255) / 256, 256, 0, stream>>>(
            W, Wt, log2in, outd);
    };
    tp(fc1_w, wt_fc1, 7, 128);
    tp(fc2_w, wt_fc2, 7, 128);
    tp(fc3_w, wt_fc3, 8, 256);
    tp(fc4_w, wt_fc4, 9, 512);
    tp(fc_w,  wt_fc,  9, 256);
    tp(w_ih,                    wt_ih1, 8, 1024);
    tp(w_ih + (size_t)256*1024, wt_ih2, 8, 1024);
    tp(w_hh,                    wt_hh1, 8, 1024);
    tp(w_hh + (size_t)256*1024, wt_hh2, 8, 1024);
    tp(a1_w, wt_a1, 8, 64);
    tp(c1_w, wt_c1, 8, 64);

    // ---- embed MLP over all T*N rows
    embed_kernel<<<TN / 16, 256, 0, stream>>>(
        obs, ln0_g, ln0_b, ln1_g, ln1_b, ln2_g, ln2_b, ln3_g, ln3_b, ln4_g, ln4_b,
        fc1_b, fc2_b, fc3_b, fc4_b, fc_b,
        wt_fc1, wt_fc2, wt_fc3, wt_fc4, wt_fc, core_in);

    // ---- LSTM scan: 2 launches per step (layer1, then layer2)
    const float* b_ih0 = b_ih,        * b_hh0 = b_hh;
    const float* b_ih1 = b_ih + 1024, * b_hh1 = b_hh + 1024;
    for (int t = 0; t < TT; ++t) {
        const float* mt = masks + (size_t)t * NN;
        lstm_layer_kernel<<<NN / 16, 256, 0, stream>>>(
            core_in + (size_t)t * NN * HH, mt, h1, c1,
            wt_ih1, wt_hh1, b_ih0, b_hh0, h1bf);
        lstm_layer_kernel<<<NN / 16, 256, 0, stream>>>(
            h1bf, mt, h2, c2,
            wt_ih2, wt_hh2, b_ih1, b_hh1, core_out + (size_t)t * NN * HH);
    }

    // ---- heads
    head_kernel<<<TN / 16, 256, 0, stream>>>(
        core_out, wt_a1, wt_c1, a1_b, c1_b, a2_w, a2_b, c2_w, c2_b, logits, value);
}
